// RelativeGlobalAttention_39599598469319
// MI455X (gfx1250) — compile-verified, matrix operand order checked by reading, not run on hardware
//
#include <hip/hip_runtime.h>
#include <hip/hip_bf16.h>

#define DEV static __device__ __forceinline__

typedef __attribute__((ext_vector_type(16))) __bf16 v16bf;
typedef __attribute__((ext_vector_type(8)))  float  v8f;

union FragCast { uint4 u[2]; v16bf v; };

DEV unsigned short f2bf(float f) {
  union { float f; unsigned u; } v; v.f = f;
  unsigned r = v.u + 0x7fffu + ((v.u >> 16) & 1u);
  return (unsigned short)(r >> 16);
}
DEV float bf2f(unsigned short b) {
  union { unsigned u; float f; } v; v.u = ((unsigned)b) << 16;
  return v.f;
}

// Load a 16x32 bf16 operand fragment (A-style layout; also used for B where
// "row0" = output-column base and the source is stored column-of-B-contiguous).
// Per ISA 7.12.2: lanes 0-15 hold K {k0..k0+7, k0+16..k0+23},
//                 lanes 16-31 hold K {k0+8..k0+15, k0+24..k0+31}.
DEV v16bf load_frag(const unsigned short* p, int stride, int row0, int k0) {
  int lane = threadIdx.x & 31;
  int lr = lane & 15, hf = lane >> 4;
  const unsigned short* r = p + (size_t)(row0 + lr) * stride + k0 + hf * 8;
  FragCast f;
  f.u[0] = *(const uint4*)(r);
  f.u[1] = *(const uint4*)(r + 16);
  return f.v;
}

DEV v8f wmma_bf16(v16bf a, v16bf b, v8f c) {
  return __builtin_amdgcn_wmma_f32_16x16x32_bf16(false, a, false, b,
                                                 (short)0, c, false, false);
}

__global__ void f32_to_bf16_k(const float* __restrict__ in,
                              unsigned short* __restrict__ out, int n) {
  int i = blockIdx.x * blockDim.x + threadIdx.x;
  int st = gridDim.x * blockDim.x;
  for (; i < n; i += st) out[i] = f2bf(in[i]);
}

// C[M,N] = A[M,K] * B^T where Bt is stored [N,K] row-major (i.e. B column n
// is contiguous in k). mode: 0 = bf16 row-major store, 1 = bf16 store
// transposed per head (V -> Vt[b,h,dh,l]), 2 = f32 row-major store (+bias).
__global__ __launch_bounds__(256)
void gemm_bf16_wmma(const unsigned short* __restrict__ A,
                    const unsigned short* __restrict__ Bt,
                    const float* __restrict__ bias,
                    void* __restrict__ out,
                    int M, int N, int Kd, int mode) {
  int wave = threadIdx.x >> 5;
  int lane = threadIdx.x & 31;
  int lr = lane & 15, hf = lane >> 4;
  int rowBase = blockIdx.y * 128 + (wave >> 1) * 32;  // 32 rows per wave
  int colBase = blockIdx.x * 128 + (wave & 1) * 64;   // 64 cols per wave
  v8f acc[2][4];
#pragma unroll
  for (int i = 0; i < 2; i++)
#pragma unroll
    for (int j = 0; j < 4; j++)
#pragma unroll
      for (int e = 0; e < 8; e++) acc[i][j][e] = 0.0f;

  for (int k0 = 0; k0 < Kd; k0 += 32) {
    v16bf a0 = load_frag(A, Kd, rowBase, k0);
    v16bf a1 = load_frag(A, Kd, rowBase + 16, k0);
#pragma unroll
    for (int j = 0; j < 4; j++) {
      v16bf bq = load_frag(Bt, Kd, colBase + 16 * j, k0);
      acc[0][j] = wmma_bf16(a0, bq, acc[0][j]);
      acc[1][j] = wmma_bf16(a1, bq, acc[1][j]);
    }
  }

#pragma unroll
  for (int i = 0; i < 2; i++)
#pragma unroll
    for (int j = 0; j < 4; j++) {
      int col = colBase + 16 * j + lr;
      float bv = bias ? bias[col] : 0.0f;
#pragma unroll
      for (int e = 0; e < 8; e++) {
        int row = rowBase + 16 * i + e + hf * 8;
        float v = acc[i][j][e] + bv;
        if (mode == 0) {
          ((unsigned short*)out)[(size_t)row * N + col] = f2bf(v);
        } else if (mode == 1) {
          int b = row >> 11, l = row & 2047;   // L = 2048
          int h = col >> 6,  dh = col & 63;    // DH = 64
          ((unsigned short*)out)[((size_t)((b * 16 + h) * 64 + dh) << 11) + l] = f2bf(v);
        } else {
          ((float*)out)[(size_t)row * N + col] = v;
        }
      }
    }
}

#define LSEQ 2048

// One wave per 16-row block per (b,h). Srel[l,m] = q_l . E[2047+m-l] for m<=l.
// Templated on QE-strip width W (LDS bytes = 32*W): blocks with l0 <= W-16
// only ever touch e >= 2048-W, so the strip is stored with column offset
// EOFF = 2048-W. W=1024 halves LDS -> ~2x workgroups resident per WGP.
template <int W>
__global__ __launch_bounds__(32)
void rga_attention(const unsigned short* __restrict__ Q,
                   const unsigned short* __restrict__ Kb,
                   const unsigned short* __restrict__ Vt,
                   const unsigned short* __restrict__ Eb,
                   unsigned short* __restrict__ O, int lb_base) {
  __shared__ unsigned short qe[16 * W];                    // bf16 QE strip
  __shared__ __align__(16) unsigned short pstage[16 * 32]; // P tile staging
  const int EOFF = LSEQ - W;

  int lane = threadIdx.x & 31;
  int lr = lane & 15, hf = lane >> 4;
  int lb = lb_base + (blockIdx.x & 63);
  int bh = blockIdx.x >> 6;
  int h = bh & 15, b = bh >> 4;
  int l0 = lb << 4;
  int qrow0 = b * LSEQ + l0;
  int hk = h * 64;

  v16bf qa0 = load_frag(Q, 1024, qrow0, hk);
  v16bf qa1 = load_frag(Q, 1024, qrow0, hk + 32);

  // ---- QE strip: only e >= 2047-(l0+15) is ever read ----
  int ebs = (2032 - l0) >> 4;
  for (int eb = ebs; eb < 128; eb++) {
    int e0 = eb << 4;
    v8f acc;
#pragma unroll
    for (int e = 0; e < 8; e++) acc[e] = 0.0f;
    acc = wmma_bf16(qa0, load_frag(Eb, 64, e0, 0), acc);
    acc = wmma_bf16(qa1, load_frag(Eb, 64, e0, 32), acc);
#pragma unroll
    for (int e = 0; e < 8; e++)
      qe[(e + hf * 8) * W + e0 - EOFF + lr] = f2bf(acc[e]);
  }
  __syncthreads();

  float rmax[8], rsum[8];
  v8f oacc[4];
#pragma unroll
  for (int j = 0; j < 8; j++) { rmax[j] = -3.0e38f; rsum[j] = 0.0f; }
#pragma unroll
  for (int t = 0; t < 4; t++)
#pragma unroll
    for (int j = 0; j < 8; j++) oacc[t][j] = 0.0f;

  int krow0 = b * LSEQ;
  int vrow0 = (b * 16 + h) * 64;
  const float inv = 0.125f;  // 1/sqrt(DH)

  for (int m0 = 0; m0 < LSEQ; m0 += 32) {
    // Prefetch next iteration's K rows (one 128B head-slice cacheline per
    // lane) and Vt rows into near caches: global_prefetch_b8.
    int m0n = m0 + 32;
    if (m0n < LSEQ) {
      __builtin_prefetch(Kb + (size_t)(krow0 + m0n + lane) * 1024 + hk, 0, 3);
      __builtin_prefetch(Vt + (size_t)(vrow0 + lane) * LSEQ + m0n, 0, 3);
      __builtin_prefetch(Vt + (size_t)(vrow0 + 32 + lane) * LSEQ + m0n, 0, 3);
    }

    v8f s0, s1;
#pragma unroll
    for (int j = 0; j < 8; j++) { s0[j] = 0.0f; s1[j] = 0.0f; }
    s0 = wmma_bf16(qa0, load_frag(Kb, 1024, krow0 + m0, hk), s0);
    s0 = wmma_bf16(qa1, load_frag(Kb, 1024, krow0 + m0, hk + 32), s0);
    s1 = wmma_bf16(qa0, load_frag(Kb, 1024, krow0 + m0 + 16, hk), s1);
    s1 = wmma_bf16(qa1, load_frag(Kb, 1024, krow0 + m0 + 16, hk + 32), s1);

#pragma unroll
    for (int j = 0; j < 8; j++) {
      int rt = j + hf * 8;
      int row = l0 + rt;
      int ma = m0 + lr, mb = m0 + 16 + lr;
      float v0 = s0[j], v1 = s1[j];
      if (ma <= row) v0 += bf2f(qe[rt * W + 2047 + ma - row - EOFF]);
      if (mb <= row) v1 += bf2f(qe[rt * W + 2047 + mb - row - EOFF]);
      v0 *= inv; v1 *= inv;
      // row max over the 32 columns (16-lane half reduction)
      float mx = fmaxf(v0, v1);
      mx = fmaxf(mx, __shfl_xor(mx, 1, 16));
      mx = fmaxf(mx, __shfl_xor(mx, 2, 16));
      mx = fmaxf(mx, __shfl_xor(mx, 4, 16));
      mx = fmaxf(mx, __shfl_xor(mx, 8, 16));
      float nm = fmaxf(rmax[j], mx);
      float corr = __expf(rmax[j] - nm);
      rmax[j] = nm;
      float p0 = __expf(v0 - nm), p1 = __expf(v1 - nm);
      float rs = p0 + p1;
      rs += __shfl_xor(rs, 1, 16);
      rs += __shfl_xor(rs, 2, 16);
      rs += __shfl_xor(rs, 4, 16);
      rs += __shfl_xor(rs, 8, 16);
      rsum[j] = rsum[j] * corr + rs;
#pragma unroll
      for (int t = 0; t < 4; t++) oacc[t][j] *= corr;
      pstage[rt * 32 + lr] = f2bf(p0);
      pstage[rt * 32 + 16 + lr] = f2bf(p1);
    }
    __syncthreads();
    v16bf pa = load_frag(pstage, 32, 0, 0);
#pragma unroll
    for (int t = 0; t < 4; t++)
      oacc[t] = wmma_bf16(pa, load_frag(Vt, LSEQ, vrow0 + 16 * t, m0), oacc[t]);
  }

#pragma unroll
  for (int j = 0; j < 8; j++) {
    float sc = 1.0f / rsum[j];
    int l = l0 + j + hf * 8;
    size_t ob = (size_t)(b * LSEQ + l) * 1024 + hk;
#pragma unroll
    for (int t = 0; t < 4; t++)
      O[ob + 16 * t + lr] = f2bf(oacc[t][j] * sc);
  }
}

extern "C" void kernel_launch(void* const* d_in, const int* in_sizes, int n_in,
                              void* d_out, int out_size, void* d_ws, size_t ws_size,
                              hipStream_t stream) {
  (void)in_sizes; (void)n_in; (void)out_size; (void)ws_size;
  const float* X    = (const float*)d_in[0];
  const float* Wq   = (const float*)d_in[1];
  const float* Wq_b = (const float*)d_in[2];
  const float* Wk   = (const float*)d_in[3];
  const float* Wk_b = (const float*)d_in[4];
  const float* Wv   = (const float*)d_in[5];
  const float* Wv_b = (const float*)d_in[6];
  const float* Wfc  = (const float*)d_in[7];
  const float* fc_b = (const float*)d_in[8];
  const float* E    = (const float*)d_in[9];

  const int NT = 4096;  // B * L
  const int D  = 1024;
  const size_t ND = (size_t)NT * D;
  const size_t DD = (size_t)D * D;
  const size_t ED = (size_t)2048 * 64;

  char* p = (char*)d_ws;
  unsigned short* Xb   = (unsigned short*)p; p += ND * 2;
  unsigned short* Wqb  = (unsigned short*)p; p += DD * 2;
  unsigned short* Wkb  = (unsigned short*)p; p += DD * 2;
  unsigned short* Wvb  = (unsigned short*)p; p += DD * 2;
  unsigned short* Wfcb = (unsigned short*)p; p += DD * 2;
  unsigned short* Ebf  = (unsigned short*)p; p += ED * 2;
  unsigned short* Qb   = (unsigned short*)p; p += ND * 2;
  unsigned short* Kbf  = (unsigned short*)p; p += ND * 2;
  unsigned short* Vtb  = (unsigned short*)p; p += ND * 2;
  unsigned short* Ob   = (unsigned short*)p; p += ND * 2;

  f32_to_bf16_k<<<1024, 256, 0, stream>>>(X,   Xb,   (int)ND);
  f32_to_bf16_k<<<1024, 256, 0, stream>>>(Wq,  Wqb,  (int)DD);
  f32_to_bf16_k<<<1024, 256, 0, stream>>>(Wk,  Wkb,  (int)DD);
  f32_to_bf16_k<<<1024, 256, 0, stream>>>(Wv,  Wvb,  (int)DD);
  f32_to_bf16_k<<<1024, 256, 0, stream>>>(Wfc, Wfcb, (int)DD);
  f32_to_bf16_k<<<256,  256, 0, stream>>>(E,   Ebf,  (int)ED);

  dim3 ggrid(D / 128, NT / 128);  // (8, 32)
  gemm_bf16_wmma<<<ggrid, 256, 0, stream>>>(Xb, Wqb, Wq_b, Qb,  NT, D, D, 0);
  gemm_bf16_wmma<<<ggrid, 256, 0, stream>>>(Xb, Wkb, Wk_b, Kbf, NT, D, D, 0);
  gemm_bf16_wmma<<<ggrid, 256, 0, stream>>>(Xb, Wvb, Wv_b, Vtb, NT, D, D, 1);

  // Row-blocks l0 <= 1008 need only e >= 1024: 32KB strip (better occupancy).
  rga_attention<1024><<<2048, 32, 0, stream>>>(Qb, Kbf, Vtb, Ebf, Ob, 0);
  rga_attention<2048><<<2048, 32, 0, stream>>>(Qb, Kbf, Vtb, Ebf, Ob, 64);

  gemm_bf16_wmma<<<ggrid, 256, 0, stream>>>(Ob, Wfcb, fc_b, d_out, NT, D, D, 2);
}
